// CNV_Cifar10_Binary_51943334478058
// MI455X (gfx1250) — 8-bit matrix operand order repaired and checked by reading, compiled, not run on hardware
//
#include <hip/hip_runtime.h>


typedef __attribute__((ext_vector_type(8))) int v8i;

#define EPSF 1e-5f

__device__ __forceinline__ signed char sgn8(float f) {
  return f > 0.f ? (signed char)1 : (f < 0.f ? (signed char)-1 : (signed char)0);
}

__global__ void conv0_kernel(const float* __restrict__ x, const float* __restrict__ w,
                             const float* __restrict__ gam, const float* __restrict__ bet,
                             const float* __restrict__ mean, const float* __restrict__ var,
                             signed char* __restrict__ a1) {
  const int total = 512 * 34 * 34 * 64;
  int idx = blockIdx.x * 256 + threadIdx.x;
  if (idx >= total) return;
  int oc = idx & 63;
  int t = idx >> 6;
  int ox = t % 34; t /= 34;
  int oy = t % 34;
  int n  = t / 34;
  float s = 0.f;
#pragma unroll
  for (int c = 0; c < 3; c++)
#pragma unroll
    for (int dy = 0; dy < 3; dy++)
#pragma unroll
      for (int dx = 0; dx < 3; dx++) {
        float wv = w[((oc * 3 + c) * 3 + dy) * 3 + dx];
        float ws = wv > 0.f ? 1.f : (wv < 0.f ? -1.f : 0.f);
        s += ws * x[((n * 3 + c) * 36 + oy + dy) * 36 + ox + dx];
      }
  float inv = gam[oc] * rsqrtf(var[oc] + EPSF);
  float f = s * inv + (bet[oc] - mean[oc] * inv);
  a1[idx] = sgn8(f);
}

template <int C, int OC>
__global__ void repack_conv(const float* __restrict__ w, int* __restrict__ wf) {
  constexpr int S = (9 * C) / 64;
  const int total = (OC / 16) * S * 32 * 8;
  int tid = blockIdx.x * 256 + threadIdx.x;
  if (tid >= total) return;
  int v = tid & 7;
  int lane = (tid >> 3) & 31;
  int s = (tid >> 8) % S;
  int nblk = tid / (S * 256);
  int g = lane >> 4;
  int n = nblk * 16 + (lane & 15);
  int kb = s * 64 + ((v >> 1) * 16) + g * 8 + (v & 1) * 4;
  unsigned pack = 0;
  for (int b = 0; b < 4; b++) {
    int k = kb + b;
    int tap = k / C, c = k % C;
    int dy = tap / 3, dx = tap % 3;
    float wv = w[(((size_t)n * C + c) * 3 + dy) * 3 + dx];
    signed char sc = wv > 0.f ? 1 : (wv < 0.f ? -1 : 0);
    pack |= ((unsigned)(unsigned char)sc) << (8 * b);
  }
  wf[tid] = (int)pack;
}

template <int K, int OCreal, int NP>
__global__ void repack_fc(const float* __restrict__ w, int* __restrict__ wf) {
  constexpr int S = K / 64;
  const int total = (NP / 16) * S * 32 * 8;
  int tid = blockIdx.x * 256 + threadIdx.x;
  if (tid >= total) return;
  int v = tid & 7;
  int lane = (tid >> 3) & 31;
  int s = (tid >> 8) % S;
  int nblk = tid / (S * 256);
  int g = lane >> 4;
  int n = nblk * 16 + (lane & 15);
  int kb = s * 64 + ((v >> 1) * 16) + g * 8 + (v & 1) * 4;
  unsigned pack = 0;
  for (int b = 0; b < 4; b++) {
    int k = kb + b;
    float wv = (n < OCreal) ? w[(size_t)n * K + k] : 0.f;
    signed char sc = wv > 0.f ? 1 : (wv < 0.f ? -1 : 0);
    pack |= ((unsigned)(unsigned char)sc) << (8 * b);
  }
  wf[tid] = (int)pack;
}

template <int C, int OC, int H, int W, int NB>
__global__ void conv_bin_wmma(const signed char* __restrict__ act,
                              const int* __restrict__ wf,
                              int* __restrict__ z) {
  constexpr int OH = H - 2, OW = W - 2, S = (9 * C) / 64;
  const int lane = threadIdx.x;
  const int g = lane >> 4, l16 = lane & 15;
  const int mbase = blockIdx.x * 32;
  const int nbase = blockIdx.y * NB;

  size_t pixbase[2];
#pragma unroll
  for (int t = 0; t < 2; t++) {
    int mpix = mbase + t * 16 + l16;
    int nloc = mpix / (OH * OW);
    int rem = mpix % (OH * OW);
    int oy = rem / OW, ox = rem % OW;
    pixbase[t] = (((size_t)nloc * H + oy) * W + ox) * C;
  }

  v8i acc[2][NB];
#pragma unroll
  for (int t = 0; t < 2; t++)
#pragma unroll
    for (int j = 0; j < NB; j++) acc[t][j] = (v8i){0, 0, 0, 0, 0, 0, 0, 0};

#pragma unroll
  for (int s = 0; s < S; s++) {
    const int tap = (s * 64) / C, coff = (s * 64) % C;
    const int dy = tap / 3, dx = tap % 3;
    v8i a[2];
#pragma unroll
    for (int t = 0; t < 2; t++) {
      const signed char* ap = act + pixbase[t] + ((size_t)dy * W + dx) * C + coff;
#pragma unroll
      for (int v = 0; v < 8; v++) {
        int koff = ((v >> 1) * 16) + g * 8 + (v & 1) * 4;
        a[t][v] = *(const int*)(ap + koff);
      }
    }
#pragma unroll
    for (int j = 0; j < NB; j++) {
      const int* wp = wf + (((size_t)(nbase + j) * S + s) * 32 + lane) * 8;
      v8i b;
#pragma unroll
      for (int v = 0; v < 8; v++) b[v] = wp[v];
      acc[0][j] = __builtin_amdgcn_wmma_i32_16x16x64_iu8(true, a[0], true, b, acc[0][j], false, false);
      acc[1][j] = __builtin_amdgcn_wmma_i32_16x16x64_iu8(true, a[1], true, b, acc[1][j], false, false);
    }
  }
#pragma unroll
  for (int t = 0; t < 2; t++)
#pragma unroll
    for (int j = 0; j < NB; j++) {
      int* zp = z + (size_t)(mbase + t * 16) * OC + (nbase + j) * 16 + l16;
#pragma unroll
      for (int r = 0; r < 8; r++) zp[(size_t)(r + g * 8) * OC] = acc[t][j][r];
    }
}

template <int K, int NP, int NB>
__global__ void gemm_bin_wmma(const signed char* __restrict__ act,
                              const int* __restrict__ wf,
                              int* __restrict__ z) {
  constexpr int S = K / 64;
  const int lane = threadIdx.x, g = lane >> 4, l16 = lane & 15;
  const int mbase = blockIdx.x * 32, nbase = blockIdx.y * NB;
  const signed char* arow[2] = {act + (size_t)(mbase + l16) * K,
                                act + (size_t)(mbase + 16 + l16) * K};
  v8i acc[2][NB];
#pragma unroll
  for (int t = 0; t < 2; t++)
#pragma unroll
    for (int j = 0; j < NB; j++) acc[t][j] = (v8i){0, 0, 0, 0, 0, 0, 0, 0};

#pragma unroll
  for (int s = 0; s < S; s++) {
    v8i a[2];
#pragma unroll
    for (int t = 0; t < 2; t++)
#pragma unroll
      for (int v = 0; v < 8; v++) {
        int koff = s * 64 + ((v >> 1) * 16) + g * 8 + (v & 1) * 4;
        a[t][v] = *(const int*)(arow[t] + koff);
      }
#pragma unroll
    for (int j = 0; j < NB; j++) {
      const int* wp = wf + (((size_t)(nbase + j) * S + s) * 32 + lane) * 8;
      v8i b;
#pragma unroll
      for (int v = 0; v < 8; v++) b[v] = wp[v];
      acc[0][j] = __builtin_amdgcn_wmma_i32_16x16x64_iu8(true, a[0], true, b, acc[0][j], false, false);
      acc[1][j] = __builtin_amdgcn_wmma_i32_16x16x64_iu8(true, a[1], true, b, acc[1][j], false, false);
    }
  }
#pragma unroll
  for (int t = 0; t < 2; t++)
#pragma unroll
    for (int j = 0; j < NB; j++) {
      int* zp = z + (size_t)(mbase + t * 16) * NP + (nbase + j) * 16 + l16;
#pragma unroll
      for (int r = 0; r < 8; r++) zp[(size_t)(r + g * 8) * NP] = acc[t][j][r];
    }
}

template <int OC>
__global__ void bn_sign(const int* __restrict__ z, signed char* __restrict__ out,
                        const float* __restrict__ g, const float* __restrict__ b,
                        const float* __restrict__ m, const float* __restrict__ v,
                        int count) {
  int idx = blockIdx.x * 256 + threadIdx.x;
  if (idx >= count) return;
  int oc = idx % OC;
  float inv = g[oc] * rsqrtf(v[oc] + EPSF);
  float f = (float)z[idx] * inv + (b[oc] - m[oc] * inv);
  out[idx] = sgn8(f);
}

template <int OC>
__global__ void pool_bn_sign(const int* __restrict__ z, signed char* __restrict__ out,
                             const float* __restrict__ g, const float* __restrict__ b,
                             const float* __restrict__ m, const float* __restrict__ v,
                             int chunk, int OH, int OW) {
  int PH = OH >> 1, PW = OW >> 1;
  int total = chunk * PH * PW * OC;
  int idx = blockIdx.x * 256 + threadIdx.x;
  if (idx >= total) return;
  int oc = idx % OC;
  int t = idx / OC;
  int px = t % PW; t /= PW;
  int py = t % PH;
  int nloc = t / PH;
  const int* zp = z + (((size_t)nloc * OH + 2 * py) * OW + 2 * px) * OC + oc;
  size_t rs = (size_t)OW * OC;
  int a0 = zp[0], a1 = zp[OC], a2 = zp[rs], a3 = zp[rs + OC];
  int mx = a0 > a1 ? a0 : a1;
  int mx2 = a2 > a3 ? a2 : a3;
  if (mx2 > mx) mx = mx2;
  float inv = g[oc] * rsqrtf(v[oc] + EPSF);
  float f = (float)mx * inv + (b[oc] - m[oc] * inv);
  out[idx] = sgn8(f);
}

__global__ void final_lsm(const int* __restrict__ z, const float* __restrict__ m,
                          const float* __restrict__ v, float* __restrict__ out) {
  int n = blockIdx.x * 64 + threadIdx.x;
  if (n >= 512) return;
  float h[10];
  float mx = -1e30f;
  for (int i = 0; i < 10; i++) {
    h[i] = ((float)z[n * 16 + i] - m[i]) * rsqrtf(v[i] + EPSF);
    mx = fmaxf(mx, h[i]);
  }
  float s = 0.f;
  for (int i = 0; i < 10; i++) s += expf(h[i] - mx);
  float lse = mx + logf(s);
  for (int i = 0; i < 10; i++) out[n * 10 + i] = h[i] - lse;
}

extern "C" void kernel_launch(void* const* d_in, const int* in_sizes, int n_in,
                              void* d_out, int out_size, void* d_ws, size_t ws_size,
                              hipStream_t stream) {
  (void)in_sizes; (void)n_in; (void)out_size; (void)ws_size;

  const float* x = (const float*)d_in[0];
  const float *cw[6], *cg[6], *cb[6], *cm[6], *cv[6];
  for (int i = 0; i < 6; i++) {
    cw[i] = (const float*)d_in[1 + i];
    cg[i] = (const float*)d_in[7 + i];
    cb[i] = (const float*)d_in[13 + i];
    cm[i] = (const float*)d_in[19 + i];
    cv[i] = (const float*)d_in[25 + i];
  }
  const float* fw[3] = {(const float*)d_in[31], (const float*)d_in[32], (const float*)d_in[33]};
  const float* fg[2] = {(const float*)d_in[34], (const float*)d_in[35]};
  const float* fbt[2] = {(const float*)d_in[36], (const float*)d_in[37]};
  const float* fm[3] = {(const float*)d_in[38], (const float*)d_in[39], (const float*)d_in[40]};
  const float* fv[3] = {(const float*)d_in[41], (const float*)d_in[42], (const float*)d_in[43]};

  char* ws = (char*)d_ws;
  int* WF1  = (int*)(ws + 0x000000);
  int* WF2  = (int*)(ws + 0x010000);
  int* WF3  = (int*)(ws + 0x024000);
  int* WF4  = (int*)(ws + 0x048000);
  int* WF5  = (int*)(ws + 0x090000);
  int* WFF0 = (int*)(ws + 0x120000);
  int* WFF1 = (int*)(ws + 0x140000);
  int* WFF2 = (int*)(ws + 0x180000);
  signed char* P0 = (signed char*)(ws + (size_t)0x200000);
  signed char* P1 = (signed char*)(ws + (size_t)44040192);
  int* R          = (int*)(ws + (size_t)54525952);

  repack_conv<64, 64><<<36, 256, 0, stream>>>(cw[1], WF1);
  repack_conv<64, 128><<<72, 256, 0, stream>>>(cw[2], WF2);
  repack_conv<128, 128><<<144, 256, 0, stream>>>(cw[3], WF3);
  repack_conv<128, 256><<<288, 256, 0, stream>>>(cw[4], WF4);
  repack_conv<256, 256><<<576, 256, 0, stream>>>(cw[5], WF5);
  repack_fc<256, 512, 512><<<128, 256, 0, stream>>>(fw[0], WFF0);
  repack_fc<512, 512, 512><<<256, 256, 0, stream>>>(fw[1], WFF1);
  repack_fc<512, 10, 16><<<8, 256, 0, stream>>>(fw[2], WFF2);

  conv0_kernel<<<147968, 256, 0, stream>>>(x, cw[0], cg[0], cb[0], cm[0], cv[0], P0);

  const int CH = 64;
  for (int c0 = 0; c0 < 512; c0 += CH) {
    conv_bin_wmma<64, 64, 34, 34, 4><<<dim3((CH * 32 * 32) / 32, 1), 32, 0, stream>>>(
        P0 + (size_t)c0 * 34 * 34 * 64, WF1, R);
    pool_bn_sign<64><<<(CH * 16 * 16 * 64) / 256, 256, 0, stream>>>(
        R, P1 + (size_t)c0 * 16 * 16 * 64, cg[1], cb[1], cm[1], cv[1], CH, 32, 32);
  }
  for (int c0 = 0; c0 < 512; c0 += CH) {
    conv_bin_wmma<64, 128, 16, 16, 4><<<dim3((CH * 14 * 14) / 32, 2), 32, 0, stream>>>(
        P1 + (size_t)c0 * 16 * 16 * 64, WF2, R);
    bn_sign<128><<<(CH * 14 * 14 * 128) / 256, 256, 0, stream>>>(
        R, P0 + (size_t)c0 * 14 * 14 * 128, cg[2], cb[2], cm[2], cv[2], CH * 14 * 14 * 128);
  }
  for (int c0 = 0; c0 < 512; c0 += CH) {
    conv_bin_wmma<128, 128, 14, 14, 4><<<dim3((CH * 12 * 12) / 32, 2), 32, 0, stream>>>(
        P0 + (size_t)c0 * 14 * 14 * 128, WF3, R);
    pool_bn_sign<128><<<(CH * 6 * 6 * 128) / 256, 256, 0, stream>>>(
        R, P1 + (size_t)c0 * 6 * 6 * 128, cg[3], cb[3], cm[3], cv[3], CH, 12, 12);
  }
  for (int c0 = 0; c0 < 512; c0 += CH) {
    conv_bin_wmma<128, 256, 6, 6, 4><<<dim3((CH * 4 * 4) / 32, 4), 32, 0, stream>>>(
        P1 + (size_t)c0 * 6 * 6 * 128, WF4, R);
    bn_sign<256><<<(CH * 4 * 4 * 256) / 256, 256, 0, stream>>>(
        R, P0 + (size_t)c0 * 4 * 4 * 256, cg[4], cb[4], cm[4], cv[4], CH * 4 * 4 * 256);
  }
  for (int c0 = 0; c0 < 512; c0 += CH) {
    conv_bin_wmma<256, 256, 4, 4, 4><<<dim3((CH * 2 * 2) / 32, 4), 32, 0, stream>>>(
        P0 + (size_t)c0 * 4 * 4 * 256, WF5, R);
    pool_bn_sign<256><<<(CH * 1 * 1 * 256) / 256, 256, 0, stream>>>(
        R, P1 + (size_t)c0 * 256, cg[5], cb[5], cm[5], cv[5], CH, 2, 2);
  }

  gemm_bin_wmma<256, 512, 4><<<dim3(16, 8), 32, 0, stream>>>(P1, WFF0, R);
  bn_sign<512><<<(512 * 512) / 256, 256, 0, stream>>>(R, P0, fg[0], fbt[0], fm[0], fv[0], 512 * 512);
  gemm_bin_wmma<512, 512, 4><<<dim3(16, 8), 32, 0, stream>>>(P0, WFF1, R);
  bn_sign<512><<<(512 * 512) / 256, 256, 0, stream>>>(R, P1, fg[1], fbt[1], fm[1], fv[1], 512 * 512);
  gemm_bin_wmma<512, 16, 1><<<dim3(16, 1), 32, 0, stream>>>(P1, WFF2, R);
  final_lsm<<<8, 64, 0, stream>>>(R, fm[2], fv[2], (float*)d_out);
}
